// RNNModel_29540785062017
// MI455X (gfx1250) — matrix operand order repaired and checked by reading, not compiled, not run on hardware
//
#include <hip/hip_runtime.h>

// ---------------------------------------------------------------------------
// Types
// ---------------------------------------------------------------------------
typedef __bf16 bf16;
typedef __attribute__((ext_vector_type(16))) __bf16 v16bf;
typedef __attribute__((ext_vector_type(8)))  __bf16 v8bf;
typedef __attribute__((ext_vector_type(8)))  float  v8f;
typedef __attribute__((ext_vector_type(4)))  unsigned u32x4;
typedef __attribute__((ext_vector_type(8)))  int     i32x8;
typedef __attribute__((ext_vector_type(4)))  int     i32x4;

#define HAS_TDM __has_builtin(__builtin_amdgcn_tensor_load_to_lds)

// Problem dims
#define BB   64
#define TT   128
#define EE   256
#define HH   512
#define GG   2048   // 4*H
#define NEE  32

// ---------------------------------------------------------------------------
// WMMA fragment helpers (16x16x32 bf16, f32 accum)
// A (16x32): lane -> M=lane&15, elems 0..15 => K = k0 + 16*(lane>>4) + e.
// B (32x16): lane -> N=lane&15, elems 0..15 => K = k0 + 16*(lane>>4) + e.
// C/D (16x16): VGPR r, lane -> M = r + 8*(lane>>4), N = lane&15.
// ---------------------------------------------------------------------------
__device__ __forceinline__ v16bf cat8(v8bf lo, v8bf hi) {
  return __builtin_shufflevector(lo, hi, 0,1,2,3,4,5,6,7,8,9,10,11,12,13,14,15);
}

__device__ __forceinline__ v16bf load_a_frag(const bf16* base, int rowStride,
                                             int m0, int kcol, int lane) {
  const int row = m0 + (lane & 15);
  const int sel = (lane >> 4) << 4;
  const bf16* p = base + row * rowStride + kcol + sel;
  v8bf lo = *(const v8bf*)(p);
  v8bf hi = *(const v8bf*)(p + 8);
  return cat8(lo, hi);
}

__device__ __forceinline__ v16bf load_b_frag(const bf16* __restrict__ W, int KD,
                                             int n0, int k0, int lane) {
  const int n = n0 + (lane & 15);
  const int k = k0 + ((lane >> 4) << 4);
  return *(const v16bf*)(W + (size_t)n * KD + k);
}

__device__ __forceinline__ v8f wmma_bf16(v16bf a, v16bf b, v8f c) {
  return __builtin_amdgcn_wmma_f32_16x16x32_bf16(false, a, false, b,
                                                 (short)0, c, false, false);
}

// ---------------------------------------------------------------------------
// Tensor Data Mover: 2D tile (dim1 rows of dim0 bf16 elems) global -> LDS.
// D# per CDNA5 ISA ch8: group0 = count/lds_addr/global_addr/type=2,
// group1 = data_size=2B, tensor dims/strides, tile dims.
// ---------------------------------------------------------------------------
__device__ __forceinline__ void tdm_load_2d_bf16(unsigned ldsAddr, const void* gptr,
                                                 unsigned dim0, unsigned dim1,
                                                 unsigned stride0) {
#if HAS_TDM
  const unsigned long long ga = (unsigned long long)gptr;
  u32x4 g0;
  g0[0] = 1u;                                                   // count=1
  g0[1] = ldsAddr;                                              // lds_addr
  g0[2] = (unsigned)ga;                                         // global[31:0]
  g0[3] = (unsigned)((ga >> 32) & 0x01ffffffull) | (2u << 30);  // global[56:32]|type=2
  i32x8 g1;
  g1[0] = (int)(1u << 16);                                      // data_size=1 (2B)
  g1[1] = (int)((dim0 & 0xffffu) << 16);                        // tensor_dim0 lo
  g1[2] = (int)(((dim0 >> 16) & 0xffffu) | ((dim1 & 0xffffu) << 16));
  g1[3] = (int)(((dim1 >> 16) & 0xffffu) | ((dim0 & 0xffffu) << 16)); // tile_dim0
  g1[4] = (int)(dim1 & 0xffffu);                                // tile_dim1
  g1[5] = (int)stride0;                                         // dim0 stride lo32
  g1[6] = 0;
  g1[7] = 0;
  const i32x4 z4 = (i32x4){0, 0, 0, 0};
#if defined(__clang_major__) && (__clang_major__ >= 23)
  const i32x8 z8 = (i32x8){0, 0, 0, 0, 0, 0, 0, 0};
  __builtin_amdgcn_tensor_load_to_lds(g0, g1, z4, z4, z8, 0);
#else
  __builtin_amdgcn_tensor_load_to_lds(g0, g1, z4, z4, 0);
#endif
#else
  (void)ldsAddr; (void)gptr; (void)dim0; (void)dim1; (void)stride0;
#endif
}

// ---------------------------------------------------------------------------
// Fast activations
// ---------------------------------------------------------------------------
__device__ __forceinline__ float sigm(float x) { return 1.0f / (1.0f + __expf(-x)); }
__device__ __forceinline__ float tanh_f(float x) { return 2.0f / (1.0f + __expf(-2.0f * x)) - 1.0f; }

// ---------------------------------------------------------------------------
// K1: embedding gather + f32 -> bf16, output time-major [T][B][E]
// ---------------------------------------------------------------------------
__global__ void embed_kernel(const int* __restrict__ queries,
                             const float* __restrict__ emb,
                             bf16* __restrict__ x) {
  const int i = blockIdx.x * blockDim.x + threadIdx.x;
  if (i >= TT * BB * EE) return;
  const int e = i % EE;
  const int b = (i / EE) % BB;
  const int t = i / (EE * BB);
  const int q = queries[b * TT + t];
  x[i] = (bf16)emb[(size_t)q * EE + e];
}

// ---------------------------------------------------------------------------
// K2: build combined weight [2048][din+512] bf16 = [Wih | Whh], bias = bih+bhh
// ---------------------------------------------------------------------------
__global__ void wconv_kernel(const float* __restrict__ Wih,
                             const float* __restrict__ Whh,
                             const float* __restrict__ bih,
                             const float* __restrict__ bhh,
                             bf16* __restrict__ Wc,
                             float* __restrict__ bias,
                             int din) {
  const int KD = din + HH;
  const int i = blockIdx.x * blockDim.x + threadIdx.x;
  if (i >= GG * KD) return;
  const int g = i / KD;
  const int k = i % KD;
  const float v = (k < din) ? Wih[(size_t)g * din + k]
                            : Whh[(size_t)g * HH + (k - din)];
  Wc[i] = (bf16)v;
  if (i < GG) bias[i] = bih[i] + bhh[i];
}

// ---------------------------------------------------------------------------
// K3: persistent bidirectional LSTM layer. grid=2 (dir), block=1024 (32 waves).
// Each wave owns 4 (m-block, j-block) pairs; computes all 4 gate tiles per
// pair so the LSTM cell update is lane-local. Cell state c lives in VGPRs
// for the whole sequence; hidden state h lives in LDS as bf16 (A-matrix
// h-columns). Time-step input tile is staged into LDS via TDM.
// ---------------------------------------------------------------------------
template <int DIN>
__global__ __launch_bounds__(1024) void lstm_layer_kernel(
    const bf16* __restrict__ xsrc,   // [T][64][DIN]
    const bf16* __restrict__ WcF,    // [2048][DIN+512] (fwd)
    const bf16* __restrict__ WcB,    // (bwd)
    const float* __restrict__ biasF,
    const float* __restrict__ biasB,
    bf16* __restrict__ hsOut,        // [T][64][1024] or nullptr
    float* __restrict__ hFinal,      // [64][2048]
    int hfBase) {
  constexpr int KD = DIN + HH;
  constexpr int KT = KD / 32;

  extern __shared__ char smem[];
  bf16* Ax = (bf16*)smem;                        // [64][DIN]
  bf16* Ah = (bf16*)(smem + (size_t)BB * DIN * 2); // [64][512]

  const int dir  = blockIdx.x;               // 0 fwd, 1 bwd
  const int wave = threadIdx.x >> 5;
  const int lane = threadIdx.x & 31;
  const bf16*  Wc   = dir ? WcB : WcF;
  const float* bias = dir ? biasB : biasF;
  const int hfColOff = hfBase + dir * HH;
  const int hsColOff = dir * HH;

  // h0 = 0
  for (int i = threadIdx.x; i < BB * HH; i += 1024) Ah[i] = (bf16)0.0f;

  // per-pair constants
  int m0[4], j0[4];
  float bI[4], bF[4], bG[4], bO[4];
#pragma unroll
  for (int p = 0; p < 4; ++p) {
    const int gp = wave * 4 + p;
    m0[p] = (gp & 3) << 4;
    j0[p] = (gp >> 2) << 4;
    const int j = j0[p] + (lane & 15);
    bI[p] = bias[0 * HH + j];
    bF[p] = bias[1 * HH + j];
    bG[p] = bias[2 * HH + j];
    bO[p] = bias[3 * HH + j];
  }
  v8f cst[4];
#pragma unroll
  for (int p = 0; p < 4; ++p) cst[p] = (v8f){0, 0, 0, 0, 0, 0, 0, 0};

  __syncthreads();

  for (int s = 0; s < TT; ++s) {
    const int t = dir ? (TT - 1 - s) : s;
    const bf16* xstep = xsrc + (size_t)t * BB * DIN;

    // stage x_t into LDS (TDM on wave 0; fallback: cooperative copy)
#if HAS_TDM
    if (wave == 0) {
      tdm_load_2d_bf16((unsigned)(unsigned long long)Ax, xstep,
                       (unsigned)DIN, (unsigned)BB, (unsigned)DIN);
#if __has_builtin(__builtin_amdgcn_s_wait_tensorcnt)
      __builtin_amdgcn_s_wait_tensorcnt(0);
#endif
    }
#else
    for (int i = threadIdx.x; i < BB * DIN; i += 1024) Ax[i] = xstep[i];
#endif
    // warm next time-step input in GL2
    if (s + 1 < TT) {
      const bf16* nxt = xsrc + (size_t)(dir ? (t - 1) : (t + 1)) * BB * DIN;
      __builtin_prefetch(nxt + ((threadIdx.x & 1023) << 4), 0, 1);
    }
    __syncthreads();

    // Gates = [x_t | h_{t-1}] @ [Wih | Whh]^T  via WMMA, then cell update.
    v8f hval[4];
#pragma unroll
    for (int p = 0; p < 4; ++p) {
      v8f aI = (v8f){0,0,0,0,0,0,0,0};
      v8f aF = aI, aG = aI, aO = aI;
      for (int kt = 0; kt < KT; ++kt) {
        const int k0 = kt * 32;
        const bf16* ab; int rs, kk;
        if (k0 < DIN) { ab = Ax; rs = DIN; kk = k0; }
        else          { ab = Ah; rs = HH;  kk = k0 - DIN; }
        const v16bf a  = load_a_frag(ab, rs, m0[p], kk, lane);
        const v16bf b0 = load_b_frag(Wc, KD, 0 * HH + j0[p], k0, lane);
        const v16bf b1 = load_b_frag(Wc, KD, 1 * HH + j0[p], k0, lane);
        const v16bf b2 = load_b_frag(Wc, KD, 2 * HH + j0[p], k0, lane);
        const v16bf b3 = load_b_frag(Wc, KD, 3 * HH + j0[p], k0, lane);
        aI = wmma_bf16(a, b0, aI);
        aF = wmma_bf16(a, b1, aF);
        aG = wmma_bf16(a, b2, aG);
        aO = wmma_bf16(a, b3, aO);
      }
      v8f hv;
#pragma unroll
      for (int r = 0; r < 8; ++r) {
        const float si = sigm(aI[r] + bI[p]);
        const float sf = sigm(aF[r] + bF[p]);
        const float so = sigm(aO[r] + bO[p]);
        const float tg = tanh_f(aG[r] + bG[p]);
        const float c  = sf * cst[p][r] + si * tg;
        cst[p][r] = c;
        hv[r] = so * tanh_f(c);
      }
      hval[p] = hv;
    }

    __syncthreads();  // all waves done reading Ah before overwriting h

#pragma unroll
    for (int p = 0; p < 4; ++p) {
      const int jj = j0[p] + (lane & 15);
      const int hi = (lane >> 4) << 3;
#pragma unroll
      for (int r = 0; r < 8; ++r) {
        const int M = m0[p] + hi + r;
        const float h = hval[p][r];
        Ah[M * HH + jj] = (bf16)h;
        if (hsOut)
          hsOut[((size_t)t * BB + M) * (size_t)(2 * HH) + hsColOff + jj] = (bf16)h;
        if (s == TT - 1)
          hFinal[(size_t)M * GG + hfColOff + jj] = h;
      }
    }
    __syncthreads();  // h visible before next step's GEMM
  }
}

// ---------------------------------------------------------------------------
// K4: logits = h @ W_out^T + b_out, then log_softmax over 32 classes.
// wave32: one wave per batch row, lane == class; shuffle reductions.
// ---------------------------------------------------------------------------
__global__ __launch_bounds__(1024) void out_logsoftmax_kernel(
    const float* __restrict__ hf, const float* __restrict__ Wout,
    const float* __restrict__ bout, float* __restrict__ out) {
  const int b = blockIdx.x * 32 + (threadIdx.x >> 5);  // 0..63
  const int lane = threadIdx.x & 31;                   // class 0..31
  const float* hrow = hf + (size_t)b * GG;
  const float* wrow = Wout + (size_t)lane * GG;
  float acc = bout[lane];
  for (int k = 0; k < GG; k += 4) {
    acc += hrow[k + 0] * wrow[k + 0] + hrow[k + 1] * wrow[k + 1] +
           hrow[k + 2] * wrow[k + 2] + hrow[k + 3] * wrow[k + 3];
  }
  float m = acc;
  for (int o = 16; o > 0; o >>= 1) m = fmaxf(m, __shfl_xor(m, o, 32));
  float sum = __expf(acc - m);
  for (int o = 16; o > 0; o >>= 1) sum += __shfl_xor(sum, o, 32);
  out[b * NEE + lane] = (acc - m) - __logf(sum);
}

// ---------------------------------------------------------------------------
// Host launcher
// ---------------------------------------------------------------------------
extern "C" void kernel_launch(void* const* d_in, const int* in_sizes, int n_in,
                              void* d_out, int out_size, void* d_ws, size_t ws_size,
                              hipStream_t stream) {
  (void)in_sizes; (void)n_in; (void)out_size; (void)ws_size;

  const int*   queries = (const int*)d_in[0];
  const float* emb     = (const float*)d_in[1];
  const float* Wih[4] = {(const float*)d_in[2],  (const float*)d_in[6],
                         (const float*)d_in[10], (const float*)d_in[14]};
  const float* Whh[4] = {(const float*)d_in[3],  (const float*)d_in[7],
                         (const float*)d_in[11], (const float*)d_in[15]};
  const float* bih[4] = {(const float*)d_in[4],  (const float*)d_in[8],
                         (const float*)d_in[12], (const float*)d_in[16]};
  const float* bhh[4] = {(const float*)d_in[5],  (const float*)d_in[9],
                         (const float*)d_in[13], (const float*)d_in[17]};
  const float* Wout = (const float*)d_in[18];
  const float* bout = (const float*)d_in[19];
  float* out = (float*)d_out;

  // workspace carve-out (256B aligned regions)
  size_t off = 0;
  auto alloc = [&](size_t bytes) -> char* {
    char* p = (char*)d_ws + off;
    off += (bytes + 255) & ~(size_t)255;
    return p;
  };
  const int din[4] = {EE, EE, 2 * HH, 2 * HH};
  bf16* xbf = (bf16*)alloc((size_t)TT * BB * EE * 2);
  bf16* Wc[4]; float* bias[4];
  for (int d = 0; d < 4; ++d) {
    Wc[d]   = (bf16*)alloc((size_t)GG * (din[d] + HH) * 2);
    bias[d] = (float*)alloc((size_t)GG * 4);
  }
  bf16*  hs0    = (bf16*)alloc((size_t)TT * BB * 2 * HH * 2);
  float* hfinal = (float*)alloc((size_t)BB * GG * 4);

  // 1) embedding -> bf16 [T][B][E]
  {
    const int total = TT * BB * EE;
    embed_kernel<<<(total + 255) / 256, 256, 0, stream>>>(queries, emb, xbf);
  }
  // 2) combined weights + bias per direction-layer
  for (int d = 0; d < 4; ++d) {
    const int total = GG * (din[d] + HH);
    wconv_kernel<<<(total + 255) / 256, 256, 0, stream>>>(
        Wih[d], Whh[d], bih[d], bhh[d], Wc[d], bias[d], din[d]);
  }
  // 3) layer 0 (din=256): writes hs0 [T][B][1024] and hfinal cols [0,1024)
  {
    const size_t smem = (size_t)BB * (EE + HH) * 2;  // 96 KB
    lstm_layer_kernel<EE><<<2, 1024, smem, stream>>>(
        xbf, Wc[0], Wc[1], bias[0], bias[1], hs0, hfinal, 0);
  }
  // 4) layer 1 (din=1024): reads hs0, writes hfinal cols [1024,2048)
  {
    const size_t smem = (size_t)BB * (2 * HH + HH) * 2;  // 192 KB
    lstm_layer_kernel<2 * HH><<<2, 1024, smem, stream>>>(
        hs0, Wc[2], Wc[3], bias[2], bias[3], nullptr, hfinal, 1024);
  }
  // 5) projection + log_softmax (64 rows = 64 waves, lane==class)
  out_logsoftmax_kernel<<<2, 1024, 0, stream>>>(hfinal, Wout, bout, out);
}
